// WESS_Encoder_10118942949694
// MI455X (gfx1250) — matrix operand order repaired and checked by reading, not compiled, not run on hardware
//
#include <hip/hip_runtime.h>

// ---------------------------------------------------------------------------
// WESS encoder for MI455X (gfx1250, wave32, WMMA bf16 16x16x32 -> f32)
// GEMMs: bf16 A/W, f32 accumulate via v_wmma_f32_16x16x32_bf16.
// Tile staging: GLOBAL_LOAD_ASYNC_TO_LDS_B128 double-buffered, s_wait_asynccnt.
// ---------------------------------------------------------------------------

typedef __attribute__((ext_vector_type(16))) __bf16 v16bf;
typedef __attribute__((ext_vector_type(8)))  __bf16 v8bf;
typedef __attribute__((ext_vector_type(8)))  float  v8f;

#define DV 768
#define G3 2304

__device__ __forceinline__ __bf16 f2bf(float f) {
  unsigned u = __float_as_uint(f);
  u += 0x7fffu + ((u >> 16) & 1u);          // round-to-nearest-even
  unsigned short h = (unsigned short)(u >> 16);
  __bf16 r; __builtin_memcpy(&r, &h, 2); return r;
}
__device__ __forceinline__ float bf2f(__bf16 b) {
  unsigned short h; __builtin_memcpy(&h, &b, 2);
  return __uint_as_float(((unsigned)h) << 16);
}
__device__ __forceinline__ float sinus_pe(int pos, int i) {
  float ex = (float)(2 * (i >> 1)) * (1.0f / 768.0f);
  float ang = (float)pos * __powf(10000.0f, -ex);
  return (i & 1) ? __cosf(ang) : __sinf(ang);
}
__device__ __forceinline__ float sigm(float x) { return 1.0f / (1.0f + __expf(-x)); }
__device__ __forceinline__ float gelu_tanh(float x) {
  return 0.5f * x * (1.0f + tanhf(0.7978845608028654f * (x + 0.044715f * x * x * x)));
}

// Async copy of 16 bytes per lane: global -> LDS, tracked by ASYNCcnt.
__device__ __forceinline__ void async_copy16(unsigned lds_off, const void* g) {
  asm volatile("global_load_async_to_lds_b128 %0, %1, off"
               :: "v"(lds_off), "v"(g) : "memory");
}

// ---------------------------------------------------------------------------
// Generic WMMA GEMM:  C[M,N] = act(alpha * A[M,K] @ op(W) + bias[N]) (+ C)
//   TRANS_W=1 : W is [N,K] row-major. Tile 128x128, waves 4(M)x2(N),
//               each wave 32x64 = 2 A-frags x 4 B-frags = 8 WMMA / K-step.
//   TRANS_W=0 : W is [K,N] row-major (att @ V). Tile 128x64, 8 waves x 16 rows.
// K staged in 32-wide slices, double-buffered in LDS via async-to-LDS;
// steady-state loop is branch-free (last tile peeled).
// Batched over grid.z with (bi,h) = (z/zdiv, z%zdiv) pointer offsets;
// bias has a per-bi stride (GRU directions carry distinct biases).
// Requires M%128==0, N%BN==0, K%32==0 (holds for every GEMM in this model).
// ---------------------------------------------------------------------------
template<bool TRANS_W, bool OUT_BF16, bool RESID, int ACT>
__global__ __launch_bounds__(256) void gemm_k(
    const __bf16* __restrict__ A, const __bf16* __restrict__ W,
    const float* __restrict__ bias, void* __restrict__ Cvoid,
    int M, int N, int K, int lda, int ldw, int ldc,
    int zdiv,
    long long sA1, long long sA2, long long sW1, long long sW2,
    long long sC1, long long sC2, long long sB1, float alpha)
{
  (void)M; (void)N;
  constexpr int BN  = TRANS_W ? 128 : 64;
  constexpr int MI  = TRANS_W ? 2 : 1;
  constexpr int BH  = TRANS_W ? 5120 : 2304;  // B region halfs per buffer
  constexpr int BUF = 5120 + BH;              // per-buffer halfs (A: 128x40)
  __shared__ __align__(16) __bf16 smem[2 * BUF];

  int z  = blockIdx.z;
  int bi = z / zdiv, hh = z - bi * zdiv;
  A += (size_t)bi * sA1 + (size_t)hh * sA2;
  W += (size_t)bi * sW1 + (size_t)hh * sW2;
  if (bias) bias += (size_t)bi * sB1;
  size_t coff = (size_t)bi * sC1 + (size_t)hh * sC2;

  int n0 = blockIdx.x * BN;
  int m0 = blockIdx.y * 128;
  int tid = threadIdx.x;
  int wave = tid >> 5;
  int lane = tid & 31;
  int mbase = TRANS_W ? ((wave >> 1) * 32) : (wave * 16);
  int nbase = TRANS_W ? ((wave & 1) * 64) : 0;

  v8f acc[MI][4];
#pragma unroll
  for (int i = 0; i < MI; ++i)
#pragma unroll
    for (int j = 0; j < 4; ++j)
#pragma unroll
      for (int e = 0; e < 8; ++e) acc[i][j][e] = 0.0f;

  // Stage one 128x32 A tile + B tile into LDS buffer `buf`.
  // Async instructions per wave per tile: TRANS 4 (A2+B2), NT 3 (A2+B1).
  auto issue_tile = [&](int k0, int buf) {
    __bf16* As = smem + buf * BUF;
    __bf16* Bs = As + 5120;
    {
      int r = tid >> 1, sg = (tid & 1) * 16;
      const __bf16* src = A + (size_t)(m0 + r) * lda + k0 + sg;
      unsigned dst = (unsigned)(size_t)(void*)&As[r * 40 + sg];
      async_copy16(dst, src);
      async_copy16(dst + 16, src + 8);
    }
    if (TRANS_W) { // 128x32 as [n][k]
      int r = tid >> 1, sg = (tid & 1) * 16;
      const __bf16* src = W + (size_t)(n0 + r) * ldw + k0 + sg;
      unsigned dst = (unsigned)(size_t)(void*)&Bs[r * 40 + sg];
      async_copy16(dst, src);
      async_copy16(dst + 16, src + 8);
    } else {       // 32x64 as [k][n]
      int r = tid >> 3, sg = (tid & 7) * 8;
      async_copy16((unsigned)(size_t)(void*)&Bs[r * 72 + sg],
                   W + (size_t)(k0 + r) * ldw + n0 + sg);
    }
  };

  // Consume buffer `buf`: preload all fragments, then burst the WMMAs.
  auto compute = [&](int buf) {
    const __bf16* As = smem + buf * BUF;
    const __bf16* Bs = As + 5120;
    v16bf af[MI];
#pragma unroll
    for (int i = 0; i < MI; ++i) {
      int m  = mbase + i * 16 + (lane & 15);
      int ko = (lane >> 4) * 16;
      const __bf16* pa = &As[m * 40 + ko];
      v8bf alo = *(const v8bf*)pa;
      v8bf ahi = *(const v8bf*)(pa + 8);
#pragma unroll
      for (int e = 0; e < 8; ++e) { af[i][e] = alo[e]; af[i][8 + e] = ahi[e]; }
    }
    v16bf bf[4];
#pragma unroll
    for (int j = 0; j < 4; ++j) {
      int n = nbase + j * 16 + (lane & 15);
      if (TRANS_W) {
        int ko = (lane >> 4) * 16;
        const __bf16* pb = &Bs[n * 40 + ko];
        v8bf b0 = *(const v8bf*)pb;
        v8bf b1 = *(const v8bf*)(pb + 8);
#pragma unroll
        for (int e = 0; e < 8; ++e) { bf[j][e] = b0[e]; bf[j][8 + e] = b1[e]; }
      } else {
        int kb = (lane >> 4) * 16;
#pragma unroll
        for (int e = 0; e < 16; ++e) bf[j][e] = Bs[(kb + e) * 72 + n];
      }
    }
#pragma unroll
    for (int i = 0; i < MI; ++i)
#pragma unroll
      for (int j = 0; j < 4; ++j)
        acc[i][j] = __builtin_amdgcn_wmma_f32_16x16x32_bf16(
            false, af[i], false, bf[j], (short)0, acc[i][j], false, false);
  };

  issue_tile(0, 0);
  int cur = 0;
  int k0 = 0;
  for (; k0 + 32 < K; k0 += 32) {       // branch-free steady state
    issue_tile(k0 + 32, cur ^ 1);
    if (TRANS_W) asm volatile("s_wait_asynccnt 0x4" ::: "memory");
    else         asm volatile("s_wait_asynccnt 0x3" ::: "memory");
    __syncthreads();                     // current tile visible to all waves
    compute(cur);
    __syncthreads();                     // reads done before buffer reuse
    cur ^= 1;
  }
  asm volatile("s_wait_asynccnt 0x0" ::: "memory");
  __syncthreads();
  compute(cur);                          // peeled final tile

  // Epilogue. C layout: lane n = lane&15, M = (lane>>4)*8 + vgpr
  int nn = lane & 15;
#pragma unroll
  for (int i = 0; i < MI; ++i) {
    int gm = m0 + mbase + i * 16 + (lane >> 4) * 8;
#pragma unroll
    for (int j = 0; j < 4; ++j) {
      int gn = n0 + nbase + j * 16 + nn;
      float bv = bias ? bias[gn] : 0.0f;
#pragma unroll
      for (int v = 0; v < 8; ++v) {
        float val = acc[i][j][v] * alpha + bv;
        if (ACT == 1) val = gelu_tanh(val);
        size_t idx = (size_t)(gm + v) * ldc + gn;
        if (OUT_BF16) {
          ((__bf16*)Cvoid + coff)[idx] = f2bf(val);
        } else {
          float* cp = (float*)Cvoid + coff;
          if (RESID) val += cp[idx];
          cp[idx] = val;
        }
      }
    }
  }
}

// ---------------------------------------------------------------------------
// Elementwise / small kernels
// ---------------------------------------------------------------------------
__global__ void f2bf_k(const float* __restrict__ s, __bf16* __restrict__ d, int n) {
  int i = blockIdx.x * 256 + threadIdx.x;
  if (i < n) d[i] = f2bf(s[i]);
}
__global__ void zero_f32_k(float* p, int n) {
  int i = blockIdx.x * 256 + threadIdx.x;
  if (i < n) p[i] = 0.0f;
}
__global__ void zero_bf16_k(__bf16* p, int n) {
  int i = blockIdx.x * 256 + threadIdx.x;
  if (i < n) p[i] = f2bf(0.0f);
}

// seg[tok, j] = emb[x[tok], j] + PE(tok%8, j)   (16384 x 768, bf16)
__global__ void embed_k(const int* __restrict__ x, const float* __restrict__ emb,
                        __bf16* __restrict__ seg) {
  int idx = blockIdx.x * 256 + threadIdx.x;
  int tok = idx / 768, j = idx - tok * 768;
  float v = emb[(size_t)x[tok] * 768 + j] + sinus_pe(tok & 7, j);
  seg[idx] = f2bf(v);
}

// Fused GRU gate update for BOTH directions (PyTorch gate order r,z,n).
// Buffers are [2][...] with direction-major layout. s1 = step index 0..7.
__global__ void gru_gate_k(const __bf16* __restrict__ gi, const float* __restrict__ gh,
                           const __bf16* __restrict__ hprev, __bf16* __restrict__ hnext,
                           __bf16* __restrict__ yout, float* __restrict__ weacc, int s1) {
  int idx = blockIdx.x * 256 + threadIdx.x;   // < 2*2048*768
  int dir = (idx >= 2048 * 768) ? 1 : 0;
  int loc = idx - dir * (2048 * 768);
  int r = loc / 768, j = loc - r * 768;
  int t = dir ? (7 - s1) : s1;
  const __bf16* giD = gi + (size_t)dir * 16384 * G3;
  const float*  ghD = gh + (size_t)dir * 2048 * G3;
  size_t girow = ((size_t)r * 8 + t) * G3;
  size_t ghrow = (size_t)r * G3;
  float rr = sigm(bf2f(giD[girow + j])         + ghD[ghrow + j]);
  float zz = sigm(bf2f(giD[girow + 768 + j])   + ghD[ghrow + 768 + j]);
  float nv = tanhf(bf2f(giD[girow + 1536 + j]) + rr * ghD[ghrow + 1536 + j]);
  float hv = (1.0f - zz) * nv + zz * bf2f(hprev[idx]);
  hnext[idx] = f2bf(hv);
  if (yout)  yout[((size_t)r * 8 + t) * 1536 + (size_t)dir * 768 + j] = f2bf(hv);
  if (weacc) atomicAdd(&weacc[loc], hv); // both dirs hit the same element
}

// h = we + bert_embeddings + PE(word, j)   (writes f32 into d_out)
__global__ void add3_k(const float* __restrict__ we, const float* __restrict__ bemb,
                       float* __restrict__ out) {
  int idx = blockIdx.x * 256 + threadIdx.x;
  int r = idx / 768, j = idx - r * 768;
  out[idx] = we[idx] + bemb[idx] + sinus_pe(r & 255, j);
}

// LayerNorm over 768 (one block of 256 threads per row), bf16 out
__global__ __launch_bounds__(256) void layernorm_k(const float* __restrict__ x,
    const float* __restrict__ g, const float* __restrict__ b, __bf16* __restrict__ out) {
  int row = blockIdx.x, t = threadIdx.x;
  const float* xr = x + (size_t)row * 768;
  float v0 = xr[t], v1 = xr[t + 256], v2 = xr[t + 512];
  __shared__ float s1[256], s2[256];
  s1[t] = v0 + v1 + v2;
  s2[t] = v0 * v0 + v1 * v1 + v2 * v2;
  __syncthreads();
  for (int st = 128; st > 0; st >>= 1) {
    if (t < st) { s1[t] += s1[t + st]; s2[t] += s2[t + st]; }
    __syncthreads();
  }
  float mean = s1[0] * (1.0f / 768.0f);
  float var  = s2[0] * (1.0f / 768.0f) - mean * mean;
  float inv  = rsqrtf(var + 1e-6f);
  __bf16* orow = out + (size_t)row * 768;
  orow[t]       = f2bf((v0 - mean) * inv * g[t]       + b[t]);
  orow[t + 256] = f2bf((v1 - mean) * inv * g[t + 256] + b[t + 256]);
  orow[t + 512] = f2bf((v2 - mean) * inv * g[t + 512] + b[t + 512]);
}

// Softmax over 256 (one block per row), bf16 probs out
__global__ __launch_bounds__(256) void softmax_k(const float* __restrict__ s,
                                                 __bf16* __restrict__ p) {
  int row = blockIdx.x, t = threadIdx.x;
  float v = s[(size_t)row * 256 + t];
  __shared__ float red[256];
  red[t] = v; __syncthreads();
  for (int st = 128; st > 0; st >>= 1) {
    if (t < st) red[t] = fmaxf(red[t], red[t + st]);
    __syncthreads();
  }
  float mx = red[0]; __syncthreads();
  float e = __expf(v - mx);
  red[t] = e; __syncthreads();
  for (int st = 128; st > 0; st >>= 1) {
    if (t < st) red[t] += red[t + st];
    __syncthreads();
  }
  p[(size_t)row * 256 + t] = f2bf(e / red[0]);
}

// ---------------------------------------------------------------------------
// Host orchestration
// ---------------------------------------------------------------------------
extern "C" void kernel_launch(void* const* d_in, const int* in_sizes, int n_in,
                              void* d_out, int out_size, void* d_ws, size_t ws_size,
                              hipStream_t stream) {
  (void)in_sizes; (void)n_in; (void)out_size; (void)ws_size;
  const int NTOK = 16384;   // B*S
  const int MW   = 2048;    // B*NW rows
  const float ATT_SCALE = 0.07216878364870322f; // 1/sqrt(192)

  const int*   x     = (const int*)d_in[0];
  const float* bemb  = (const float*)d_in[2];
  const float* embt  = (const float*)d_in[3];
  const float* wih0f = (const float*)d_in[4];
  const float* whh0f = (const float*)d_in[5];
  const float* bih0  = (const float*)d_in[6];
  const float* bhh0  = (const float*)d_in[7];
  const float* wih1f = (const float*)d_in[8];
  const float* whh1f = (const float*)d_in[9];
  const float* bih1  = (const float*)d_in[10];
  const float* bhh1  = (const float*)d_in[11];
  const float* wqkvf = (const float*)d_in[12];
  const float* bqkv  = (const float*)d_in[13];
  const float* wof   = (const float*)d_in[14];
  const float* bo    = (const float*)d_in[15];
  const float* w1f   = (const float*)d_in[16];
  const float* b1    = (const float*)d_in[17];
  const float* w2f   = (const float*)d_in[18];
  const float* b2    = (const float*)d_in[19];
  const float* ln1g  = (const float*)d_in[20];
  const float* ln1b  = (const float*)d_in[21];
  const float* ln2g  = (const float*)d_in[22];
  const float* ln2b  = (const float*)d_in[23];

  size_t off = 0;
  auto alloc = [&](size_t elems, size_t esz) -> void* {
    void* p = (char*)d_ws + off;
    off += (elems * esz + 255) & ~(size_t)255;
    return p;
  };
  __bf16* wWih0 = (__bf16*)alloc((size_t)2 * G3 * DV, 2);
  __bf16* wWhh0 = (__bf16*)alloc((size_t)2 * G3 * DV, 2);
  __bf16* wWih1 = (__bf16*)alloc((size_t)2 * G3 * 2 * DV, 2);
  __bf16* wWhh1 = (__bf16*)alloc((size_t)2 * G3 * DV, 2);
  __bf16* wQkv  = (__bf16*)alloc((size_t)3 * G3 * DV, 2);
  __bf16* wWo   = (__bf16*)alloc((size_t)3 * DV * DV, 2);
  __bf16* wW1   = (__bf16*)alloc((size_t)3 * 3072 * DV, 2);
  __bf16* wW2   = (__bf16*)alloc((size_t)3 * DV * 3072, 2);
  __bf16* seg   = (__bf16*)alloc((size_t)NTOK * DV, 2);
  __bf16* gi    = (__bf16*)alloc((size_t)2 * NTOK * G3, 2); // per-direction
  float*  gh    = (float*) alloc((size_t)2 * MW * G3, 4);   // per-direction
  __bf16* hA    = (__bf16*)alloc((size_t)2 * MW * DV, 2);   // per-direction
  __bf16* hB    = (__bf16*)alloc((size_t)2 * MW * DV, 2);
  __bf16* y0    = (__bf16*)alloc((size_t)NTOK * 2 * DV, 2);
  float*  we    = (float*) alloc((size_t)MW * DV, 4);
  __bf16* xn    = (__bf16*)alloc((size_t)MW * DV, 2);
  __bf16* qkv   = (__bf16*)alloc((size_t)MW * G3, 2);
  float*  scor  = (float*) alloc((size_t)32 * 256 * 256, 4);
  __bf16* probs = (__bf16*)alloc((size_t)32 * 256 * 256, 2);
  __bf16* obuf  = (__bf16*)alloc((size_t)MW * DV, 2);
  __bf16* ff1   = (__bf16*)alloc((size_t)MW * 3072, 2);

  dim3 blk(256);
  auto conv = [&](const float* s, __bf16* d, size_t n) {
    f2bf_k<<<dim3((unsigned)((n + 255) / 256)), blk, 0, stream>>>(s, d, (int)n);
  };
  conv(wih0f, wWih0, (size_t)2 * G3 * DV);
  conv(whh0f, wWhh0, (size_t)2 * G3 * DV);
  conv(wih1f, wWih1, (size_t)2 * G3 * 2 * DV);
  conv(whh1f, wWhh1, (size_t)2 * G3 * DV);
  conv(wqkvf, wQkv,  (size_t)3 * G3 * DV);
  conv(wof,   wWo,   (size_t)3 * DV * DV);
  conv(w1f,   wW1,   (size_t)3 * 3072 * DV);
  conv(w2f,   wW2,   (size_t)3 * DV * 3072);

  embed_k<<<dim3(NTOK * DV / 256), blk, 0, stream>>>(x, embt, seg);

  const int EW = MW * DV / 256;                 // 6144 blocks for [2048,768]
  const long long GID = (long long)NTOK * G3;   // gi per-direction stride
  const long long GHD = (long long)MW * G3;     // gh per-direction stride
  const long long HD  = (long long)MW * DV;     // h  per-direction stride

  // ---- GRU layer 0: both directions batched in grid.z ----
  gemm_k<true, true, false, 0><<<dim3(G3 / 128, NTOK / 128, 2), blk, 0, stream>>>(
      seg, wWih0, bih0, (void*)gi, NTOK, G3, DV, DV, DV, G3,
      1, 0, 0, (long long)G3 * DV, 0, GID, 0, G3, 1.0f);
  zero_bf16_k<<<dim3(2 * EW), blk, 0, stream>>>(hA, 2 * MW * DV);
  {
    __bf16* hc = hA; __bf16* hx = hB;
    for (int s1 = 0; s1 < 8; ++s1) {
      gemm_k<true, false, false, 0><<<dim3(G3 / 128, MW / 128, 2), blk, 0, stream>>>(
          hc, wWhh0, bhh0, (void*)gh, MW, G3, DV, DV, DV, G3,
          1, HD, 0, (long long)G3 * DV, 0, GHD, 0, G3, 1.0f);
      gru_gate_k<<<dim3(2 * EW), blk, 0, stream>>>(gi, gh, hc, hx, y0, nullptr, s1);
      __bf16* tmp = hc; hc = hx; hx = tmp;
    }
  }

  // ---- GRU layer 1: accumulates we = sum over {first,last} x {fwd,bwd} ----
  zero_f32_k<<<dim3(EW), blk, 0, stream>>>(we, MW * DV);
  gemm_k<true, true, false, 0><<<dim3(G3 / 128, NTOK / 128, 2), blk, 0, stream>>>(
      y0, wWih1, bih1, (void*)gi, NTOK, G3, 2 * DV, 2 * DV, 2 * DV, G3,
      1, 0, 0, (long long)G3 * 2 * DV, 0, GID, 0, G3, 1.0f);
  zero_bf16_k<<<dim3(2 * EW), blk, 0, stream>>>(hA, 2 * MW * DV);
  {
    __bf16* hc = hA; __bf16* hx = hB;
    for (int s1 = 0; s1 < 8; ++s1) {
      gemm_k<true, false, false, 0><<<dim3(G3 / 128, MW / 128, 2), blk, 0, stream>>>(
          hc, wWhh1, bhh1, (void*)gh, MW, G3, DV, DV, DV, G3,
          1, HD, 0, (long long)G3 * DV, 0, GHD, 0, G3, 1.0f);
      float* wacc = (s1 == 0 || s1 == 7) ? we : nullptr;
      gru_gate_k<<<dim3(2 * EW), blk, 0, stream>>>(gi, gh, hc, hx, nullptr, wacc, s1);
      __bf16* tmp = hc; hc = hx; hx = tmp;
    }
  }

  // ---- BERT input: h (in d_out, f32) = we + bert_embeddings + PE ----
  float* h = (float*)d_out;
  add3_k<<<dim3(EW), blk, 0, stream>>>(we, bemb, h);

  // ---- 3 pre-norm transformer layers ----
  for (int l = 0; l < 3; ++l) {
    layernorm_k<<<dim3(MW), blk, 0, stream>>>(h, ln1g + l * DV, ln1b + l * DV, xn);
    gemm_k<true, true, false, 0><<<dim3(G3 / 128, MW / 128, 1), blk, 0, stream>>>(
        xn, wQkv + (size_t)l * G3 * DV, bqkv + l * G3, (void*)qkv,
        MW, G3, DV, DV, DV, G3, 1, 0, 0, 0, 0, 0, 0, 0, 1.0f);
    // scores[z=bi*4+h][q][k] = q . k / sqrt(dh)   (A=q slice, W=k slice, [256,192])
    gemm_k<true, false, false, 0><<<dim3(256 / 128, 256 / 128, 32), blk, 0, stream>>>(
        qkv, qkv + 768, nullptr, (void*)scor,
        256, 256, 192, G3, G3, 256, 4,
        (long long)256 * G3, 192, (long long)256 * G3, 192,
        (long long)4 * 65536, 65536, 0, ATT_SCALE);
    softmax_k<<<dim3(32 * 256), blk, 0, stream>>>(scor, probs);
    // o[z][q][d] = probs @ v   (B = v slice [256,192] in [K,N] layout, BN=64)
    gemm_k<false, true, false, 0><<<dim3(192 / 64, 256 / 128, 32), blk, 0, stream>>>(
        probs, qkv + 1536, nullptr, (void*)obuf,
        256, 192, 256, 256, G3, DV, 4,
        (long long)4 * 65536, 65536, (long long)256 * G3, 192,
        (long long)256 * DV, 192, 0, 1.0f);
    // h += o @ Wo^T + bo
    gemm_k<true, false, true, 0><<<dim3(DV / 128, MW / 128, 1), blk, 0, stream>>>(
        obuf, wWo + (size_t)l * DV * DV, bo + l * DV, (void*)h,
        MW, DV, DV, DV, DV, DV, 1, 0, 0, 0, 0, 0, 0, 0, 1.0f);
    layernorm_k<<<dim3(MW), blk, 0, stream>>>(h, ln2g + l * DV, ln2b + l * DV, xn);
    // ff1 = gelu(xn @ W1^T + b1)
    gemm_k<true, true, false, 1><<<dim3(3072 / 128, MW / 128, 1), blk, 0, stream>>>(
        xn, wW1 + (size_t)l * 3072 * DV, b1 + l * 3072, (void*)ff1,
        MW, 3072, DV, DV, DV, 3072, 1, 0, 0, 0, 0, 0, 0, 0, 1.0f);
    // h += ff1 @ W2^T + b2
    gemm_k<true, false, true, 0><<<dim3(DV / 128, MW / 128, 1), blk, 0, stream>>>(
        ff1, wW2 + (size_t)l * DV * 3072, b2 + l * DV, (void*)h,
        MW, DV, 3072, 3072, 3072, DV, 1, 0, 0, 0, 0, 0, 0, 0, 1.0f);
  }
}
